// BrainAgeGATv2_4449586119195
// MI455X (gfx1250) — matrix operand order repaired and checked by reading, compiled, not run on hardware
//
#include <hip/hip_runtime.h>
#include <math.h>

#define NNODES 100000
#define NGRAPH 200
#define NHEAD  8
#define NCH    16
#define DHID   128
#define BN_EPS 1e-5f

typedef __attribute__((ext_vector_type(16))) __bf16 v16bf;
typedef __attribute__((ext_vector_type(8)))  float  v8f;

struct alignas(16) BF8 { __bf16 h[8]; };

static __device__ __forceinline__ float frelu(float v) { return v > 0.0f ? v : 0.0f; }
static __device__ __forceinline__ float lrelu02(float v) { return v > 0.0f ? v : 0.2f * v; }

static __device__ __forceinline__ void atomicMaxF(float* addr, float val) {
    unsigned int* ua = (unsigned int*)addr;
    unsigned int old = *ua;
    while (__uint_as_float(old) < val) {
        unsigned int assumed = old;
        old = atomicCAS(ua, assumed, __float_as_uint(val));
        if (old == assumed) break;
    }
}

// ---------------------------------------------------------------------------
// Weight convert: W [K,128] f32 row-major -> Wt [128,K] bf16 (transposed)
// ---------------------------------------------------------------------------
__global__ void wconv_kernel(const float* __restrict__ W, __bf16* __restrict__ Wt, int K) {
    int t = blockIdx.x * blockDim.x + threadIdx.x;
    if (t >= K * DHID) return;
    int k = t >> 7;
    int n = t & 127;
    Wt[(size_t)n * K + k] = (__bf16)W[(size_t)k * DHID + n];
}

// ---------------------------------------------------------------------------
// Embedding: h0_bf[n, 0..63] = bf16(relu(x[n,0:4] @ embW + embB))
// ---------------------------------------------------------------------------
__global__ void embed_kernel(const float* __restrict__ x, const float* __restrict__ W,
                             const float* __restrict__ b, __bf16* __restrict__ h0) {
    int t = blockIdx.x * blockDim.x + threadIdx.x;
    if (t >= NNODES * 64) return;
    int c = t & 63;
    int n = t >> 6;
    float acc = b[c];
    #pragma unroll
    for (int i = 0; i < 4; ++i) acc += x[n * 4 + i] * W[i * 64 + c];
    h0[t] = (__bf16)frelu(acc);
}

// ---------------------------------------------------------------------------
// WMMA GEMM: C[M,128] = A[M,K](bf16) @ B[K,128] + bias, B given transposed
// as Bt[128,K] bf16. One wave = one 16x16 tile; 8 waves/block = 16-row strip.
// ---------------------------------------------------------------------------
__global__ void __launch_bounds__(256)
gemm_bf16_wmma(const __bf16* __restrict__ A, const __bf16* __restrict__ Bt,
               const float* __restrict__ bias, float* __restrict__ Cm, int K) {
    const int lane = threadIdx.x & 31;
    const int wave = threadIdx.x >> 5;     // column tile 0..7
    const int half = lane >> 4;            // lane half selects K sub-range
    const int lr   = lane & 15;
    const int rowBase = blockIdx.x * 16;
    const int colBase = wave * 16;

    const __bf16* aRow = A  + (size_t)(rowBase + lr) * K;
    const __bf16* bRow = Bt + (size_t)(colBase + lr) * K;

    v8f acc = {};
    for (int k0 = 0; k0 < K; k0 += 32) {
        if (k0 + 32 < K) {  // prefetch next K chunk (global_prefetch_b8)
            __builtin_prefetch(aRow + k0 + 32, 0, 0);
            __builtin_prefetch(bRow + k0 + 32, 0, 0);
        }
        // A 16x32 bf16 fragment (ISA layout): lanes0-15 K={k0..k0+7, k0+16..k0+23}
        //                                     lanes16-31 K={k0+8..k0+15, k0+24..k0+31}
        BF8 ra0 = *(const BF8*)(aRow + k0 + (half ? 8 : 0));
        BF8 ra1 = *(const BF8*)(aRow + k0 + 16 + (half ? 8 : 0));
        // B 32x16 bf16 fragment: lane = column, lanes0-15 K={k0..k0+7, k0+16..k0+23}, lanes16-31 K={k0+8..k0+15, k0+24..k0+31}
        BF8 rb0 = *(const BF8*)(bRow + k0 + (half ? 8 : 0));
        BF8 rb1 = *(const BF8*)(bRow + k0 + 16 + (half ? 8 : 0));
        v16bf av, bv;
        #pragma unroll
        for (int i = 0; i < 8; ++i) {
            av[i] = ra0.h[i]; av[i + 8] = ra1.h[i];
            bv[i] = rb0.h[i]; bv[i + 8] = rb1.h[i];
        }
        acc = __builtin_amdgcn_wmma_f32_16x16x32_bf16(false, av, false, bv,
                                                      (short)0, acc, false, false);
    }
    const float bcol = bias[colBase + lr];
    #pragma unroll
    for (int j = 0; j < 8; ++j) {
        int row = rowBase + half * 8 + j;      // D layout: VGPR j -> M = half*8+j
        Cm[(size_t)row * DHID + colBase + lr] = acc[j] + bcol;
    }
}

// ---------------------------------------------------------------------------
// Per-layer scratch init
// ---------------------------------------------------------------------------
__global__ void init_layer_kernel(float* __restrict__ m, float* __restrict__ den,
                                  float* __restrict__ outb, float* __restrict__ bnsum,
                                  float* __restrict__ bnsq) {
    int i = blockIdx.x * blockDim.x + threadIdx.x;
    if (i < NNODES * NHEAD) { m[i] = -1e30f; den[i] = 0.0f; }
    if (i < NNODES * DHID) outb[i] = 0.0f;
    if (i < DHID) { bnsum[i] = 0.0f; bnsq[i] = 0.0f; }
}

// ---------------------------------------------------------------------------
// Edge phase (one thread per (edge, head)); self-loops appended past E
// ---------------------------------------------------------------------------
__global__ void edge_alpha_kernel(const int* __restrict__ srcA, const int* __restrict__ dstA,
                                  int E, int Etot, const float* __restrict__ xl,
                                  const float* __restrict__ xr, const float* __restrict__ att,
                                  float* __restrict__ alpha, float* __restrict__ m) {
    int t = blockIdx.x * blockDim.x + threadIdx.x;
    if (t >= Etot * NHEAD) return;
    int h = t & 7, e = t >> 3;
    int s, d;
    if (e < E) { s = srcA[e]; d = dstA[e]; } else { s = d = e - E; }
    const float* pl = xl + (size_t)s * DHID + h * NCH;
    const float* pr = xr + (size_t)d * DHID + h * NCH;
    const float* pa = att + h * NCH;
    float acc = 0.0f;
    #pragma unroll
    for (int c = 0; c < NCH; ++c) acc += lrelu02(pl[c] + pr[c]) * pa[c];
    alpha[t] = acc;
    atomicMaxF(&m[d * NHEAD + h], acc);
}

__global__ void edge_exp_kernel(const int* __restrict__ dstA, int E, int Etot,
                                const float* __restrict__ m, float* __restrict__ alpha,
                                float* __restrict__ den) {
    int t = blockIdx.x * blockDim.x + threadIdx.x;
    if (t >= Etot * NHEAD) return;
    int h = t & 7, e = t >> 3;
    int d = (e < E) ? dstA[e] : (e - E);
    float a = __expf(alpha[t] - m[d * NHEAD + h]);
    alpha[t] = a;
    atomicAdd(&den[d * NHEAD + h], a);
}

__global__ void edge_agg_kernel(const int* __restrict__ srcA, const int* __restrict__ dstA,
                                int E, int Etot, const float* __restrict__ xl,
                                const float* __restrict__ alpha, const float* __restrict__ den,
                                float* __restrict__ outb) {
    int t = blockIdx.x * blockDim.x + threadIdx.x;
    if (t >= Etot * NHEAD) return;
    int h = t & 7, e = t >> 3;
    int s, d;
    if (e < E) { s = srcA[e]; d = dstA[e]; } else { s = d = e - E; }
    float w = alpha[t] / den[d * NHEAD + h];
    const float* pl = xl + (size_t)s * DHID + h * NCH;
    float* po = outb + (size_t)d * DHID + h * NCH;
    #pragma unroll
    for (int c = 0; c < NCH; ++c) atomicAdd(&po[c], pl[c] * w);
}

// ---------------------------------------------------------------------------
// BatchNorm stats (two groups of 128 channels per block) + fused apply
// ---------------------------------------------------------------------------
#define BN_RPB 250
__global__ void bn_stats_kernel(const float* __restrict__ outb, float* __restrict__ bnsum,
                                float* __restrict__ bnsq) {
    int c = threadIdx.x & 127;
    int g = threadIdx.x >> 7;
    int r0 = blockIdx.x * BN_RPB;
    float s = 0.0f, q = 0.0f;
    for (int r = r0 + g; r < r0 + BN_RPB; r += 2) {
        float v = outb[(size_t)r * DHID + c];
        s += v; q += v * v;
    }
    __shared__ float ls[256], lq[256];
    ls[threadIdx.x] = s; lq[threadIdx.x] = q;
    __syncthreads();
    if (g == 0) {
        atomicAdd(&bnsum[c], ls[c] + ls[c + 128]);
        atomicAdd(&bnsq[c],  lq[c] + lq[c + 128]);
    }
}

__global__ void bn_apply_kernel(const float* __restrict__ outb, const float* __restrict__ bnsum,
                                const float* __restrict__ bnsq, const float* __restrict__ gamma,
                                const float* __restrict__ beta, float* __restrict__ hF,
                                __bf16* __restrict__ hB, int useRes) {
    int i = blockIdx.x * blockDim.x + threadIdx.x;
    if (i >= NNODES * DHID) return;
    int c = i & 127;
    float mu  = bnsum[c] * (1.0f / NNODES);
    float var = bnsq[c] * (1.0f / NNODES) - mu * mu;
    float v = (outb[i] - mu) * rsqrtf(var + BN_EPS) * gamma[c] + beta[c];
    if (useRes) v += hF[i];
    v = frelu(v);
    hF[i] = v;
    hB[i] = (__bf16)v;
}

// ---------------------------------------------------------------------------
// Global mean pool (batch is contiguous: 500 rows per graph)
// ---------------------------------------------------------------------------
__global__ void pool_kernel(const float* __restrict__ hF, float* __restrict__ pooled) {
    int g = blockIdx.x, c = threadIdx.x;
    const int rows = NNODES / NGRAPH;
    float s = 0.0f;
    size_t base = (size_t)g * rows * DHID + c;
    for (int r = 0; r < rows; ++r) s += hF[base + (size_t)r * DHID];
    pooled[g * DHID + c] = s * (1.0f / rows);
}

// ---------------------------------------------------------------------------
// Head: meta/graph/pca MLPs + 3-layer FC. One block per graph, 128 threads.
// ---------------------------------------------------------------------------
__global__ void head_kernel(const float* __restrict__ pooled, const float* __restrict__ gf,
    const float* mW1, const float* mb1, const float* mW2, const float* mb2,
    const float* gW1, const float* gb1, const float* gW2, const float* gb2,
    const float* pW1, const float* pb1, const float* pW2, const float* pb2,
    const float* fW1, const float* fb1, const float* fW2, const float* fb2,
    const float* fW3, const float* fb3, float* __restrict__ outp) {
    __shared__ float z[192], z1[128], z2[64], t1[32];
    int g = blockIdx.x, tid = threadIdx.x;
    const float* gfg = gf + g * 16;

    if (tid < 16) { float a = mb1[tid]; for (int i = 0; i < 4;  ++i) a += gfg[i]     * mW1[i*16+tid]; t1[tid] = frelu(a); }
    __syncthreads();
    if (tid < 16) { float a = mb2[tid]; for (int i = 0; i < 16; ++i) a += t1[i]      * mW2[i*16+tid]; z[128+tid] = frelu(a); }
    __syncthreads();
    if (tid < 16) { float a = gb1[tid]; for (int i = 0; i < 2;  ++i) a += gfg[4+i]   * gW1[i*16+tid]; t1[tid] = frelu(a); }
    __syncthreads();
    if (tid < 16) { float a = gb2[tid]; for (int i = 0; i < 16; ++i) a += t1[i]      * gW2[i*16+tid]; z[144+tid] = frelu(a); }
    __syncthreads();
    if (tid < 32) { float a = pb1[tid]; for (int i = 0; i < 10; ++i) a += gfg[6+i]   * pW1[i*32+tid]; t1[tid] = frelu(a); }
    __syncthreads();
    if (tid < 32) { float a = pb2[tid]; for (int i = 0; i < 32; ++i) a += t1[i]      * pW2[i*32+tid]; z[160+tid] = frelu(a); }
    z[tid] = pooled[g * DHID + tid];
    __syncthreads();
    { float a = fb1[tid]; for (int i = 0; i < 192; ++i) a += z[i]  * fW1[i*128+tid]; z1[tid] = frelu(a); }
    __syncthreads();
    if (tid < 64) { float a = fb2[tid]; for (int i = 0; i < 128; ++i) a += z1[i] * fW2[i*64+tid]; z2[tid] = frelu(a); }
    __syncthreads();
    if (tid == 0) { float a = fb3[0]; for (int i = 0; i < 64; ++i) a += z2[i] * fW3[i]; outp[g] = a; }
}

// ---------------------------------------------------------------------------
// Host launcher
// ---------------------------------------------------------------------------
extern "C" void kernel_launch(void* const* d_in, const int* in_sizes, int n_in,
                              void* d_out, int out_size, void* d_ws, size_t ws_size,
                              hipStream_t stream) {
    (void)n_in; (void)out_size; (void)ws_size;
    const float* x  = (const float*)d_in[0];
    const int*   ei = (const int*)d_in[1];
    const int E = in_sizes[1] / 2;
    const int Etot = E + NNODES;
    const int* srcA = ei;
    const int* dstA = ei + E;
    const float* gf = (const float*)d_in[3];
    const float* embW = (const float*)d_in[4];
    const float* embB = (const float*)d_in[5];

    // workspace carve-out
    char* wp = (char*)d_ws;
    auto carve = [&](size_t bytes) { void* p = (void*)wp; wp += (bytes + 255) & ~(size_t)255; return p; };
    float*  hF    = (float*) carve((size_t)NNODES * DHID * 4);
    __bf16* hB    = (__bf16*)carve((size_t)NNODES * DHID * 2);
    float*  xl    = (float*) carve((size_t)NNODES * DHID * 4);
    float*  xr    = (float*) carve((size_t)NNODES * DHID * 4);
    float*  outb  = (float*) carve((size_t)NNODES * DHID * 4);
    float*  alpha = (float*) carve((size_t)Etot * NHEAD * 4);
    float*  mbuf  = (float*) carve((size_t)NNODES * NHEAD * 4);
    float*  den   = (float*) carve((size_t)NNODES * NHEAD * 4);
    float*  bnsum = (float*) carve(DHID * 4);
    float*  bnsq  = (float*) carve(DHID * 4);
    float*  pooled= (float*) carve((size_t)NGRAPH * DHID * 4);
    __bf16* wtL[4]; __bf16* wtR[4];
    for (int l = 0; l < 4; ++l) {
        int K = (l == 0) ? 64 : 128;
        wtL[l] = (__bf16*)carve((size_t)DHID * K * 2);
        wtR[l] = (__bf16*)carve((size_t)DHID * K * 2);
    }

    // weight transpose + bf16 convert
    for (int l = 0; l < 4; ++l) {
        int K = (l == 0) ? 64 : 128;
        int base = 6 + l * 6;
        int nt = K * DHID;
        wconv_kernel<<<(nt + 255) / 256, 256, 0, stream>>>((const float*)d_in[base + 0], wtL[l], K);
        wconv_kernel<<<(nt + 255) / 256, 256, 0, stream>>>((const float*)d_in[base + 2], wtR[l], K);
    }

    // embedding -> bf16 A operand for layer 0
    embed_kernel<<<(NNODES * 64 + 255) / 256, 256, 0, stream>>>(x, embW, embB, hB);

    const int nEH  = Etot * NHEAD;
    const int gEH  = (nEH + 255) / 256;
    const int nND  = NNODES * DHID;
    const int gND  = (nND + 255) / 256;

    for (int l = 0; l < 4; ++l) {
        int K = (l == 0) ? 64 : 128;
        int base = 6 + l * 6;
        const float* bl   = (const float*)d_in[base + 1];
        const float* br   = (const float*)d_in[base + 3];
        const float* att  = (const float*)d_in[base + 4];
        const float* bng  = (const float*)d_in[30 + 2 * l];
        const float* bnb  = (const float*)d_in[31 + 2 * l];

        gemm_bf16_wmma<<<NNODES / 16, 256, 0, stream>>>(hB, wtL[l], bl, xl, K);
        gemm_bf16_wmma<<<NNODES / 16, 256, 0, stream>>>(hB, wtR[l], br, xr, K);

        init_layer_kernel<<<gND, 256, 0, stream>>>(mbuf, den, outb, bnsum, bnsq);
        edge_alpha_kernel<<<gEH, 256, 0, stream>>>(srcA, dstA, E, Etot, xl, xr, att, alpha, mbuf);
        edge_exp_kernel<<<gEH, 256, 0, stream>>>(dstA, E, Etot, mbuf, alpha, den);
        edge_agg_kernel<<<gEH, 256, 0, stream>>>(srcA, dstA, E, Etot, xl, alpha, den, outb);

        bn_stats_kernel<<<NNODES / BN_RPB, 256, 0, stream>>>(outb, bnsum, bnsq);
        bn_apply_kernel<<<gND, 256, 0, stream>>>(outb, bnsum, bnsq, bng, bnb, hF, hB, l > 0 ? 1 : 0);
    }

    pool_kernel<<<NGRAPH, 128, 0, stream>>>(hF, pooled);
    head_kernel<<<NGRAPH, 128, 0, stream>>>(pooled, gf,
        (const float*)d_in[38], (const float*)d_in[39], (const float*)d_in[40], (const float*)d_in[41],
        (const float*)d_in[42], (const float*)d_in[43], (const float*)d_in[44], (const float*)d_in[45],
        (const float*)d_in[46], (const float*)d_in[47], (const float*)d_in[48], (const float*)d_in[49],
        (const float*)d_in[50], (const float*)d_in[51], (const float*)d_in[52], (const float*)d_in[53],
        (const float*)d_in[54], (const float*)d_in[55], (float*)d_out);
}
